// Agent_80616536146252
// MI455X (gfx1250) — matrix operand order repaired and checked by reading, not compiled, not run on hardware
//
#include <hip/hip_runtime.h>
#include <math.h>
#include <stdint.h>

// ---------------- problem constants ----------------
#define BSZ   2560
#define AACT  200
#define DDIM  512           // LSTM hidden
#define BD    (BSZ*DDIM)    // 1,310,720
#define NEGV  (-99999.0f)

// d_out flat layout (floats), in reference return order
#define OUT_LOSS   0
#define OUT_STATE  2560
#define OUT_LP     5245440   // 2560 + 2*2*2560*512
#define OUT_AIDX   5757440   // + 2560*200
#define OUT_CREL   5760000
#define OUT_PRELIM 5762560

typedef __attribute__((ext_vector_type(16))) __bf16 v16bf;
typedef __attribute__((ext_vector_type(8)))  float  v8f;
typedef __attribute__((ext_vector_type(4)))  float  vf4;
typedef __attribute__((ext_vector_type(4)))  unsigned int v4u;
typedef __attribute__((ext_vector_type(8)))  int    v8i;
typedef __attribute__((ext_vector_type(4)))  int    v4i;

union AF { v16bf v; vf4 f[2]; };

static __device__ __forceinline__ unsigned short f2bf(float x) {
  unsigned int u = __float_as_uint(x);
  return (unsigned short)((u + 0x7FFFu + ((u >> 16) & 1u)) >> 16);
}
static __device__ __forceinline__ float sigm(float x) { return 1.0f / (1.0f + __expf(-x)); }
static __device__ __forceinline__ float rand01(unsigned int x) {
  x ^= x >> 17; x *= 0xed5ad4bbu; x ^= x >> 11; x *= 0xac4c1b51u;
  x ^= x >> 15; x *= 0x31848babu; x ^= x >> 14;
  return ((float)(x >> 8) + 0.5f) * (1.0f / 16777216.0f);
}

// 4 WMMAs against one A fragment; B fragments streamed from LDS row `lb`.
static __device__ __forceinline__ void wmma_row(const unsigned int* lb, const v16bf av, v8f acc[4]) {
#pragma unroll
  for (int t = 0; t < 4; ++t) {
    AF bfr;
    bfr.f[0] = *(const vf4*)(lb + t * 256);
    bfr.f[1] = *(const vf4*)(lb + t * 256 + 4);
    acc[t] = __builtin_amdgcn_wmma_f32_16x16x32_bf16(
        false, av, false, bfr.v, (short)0, acc[t], false, false);
  }
}

// ---------------- weight pack: f32 [K,N] -> bf16 WMMA B-fragment tiles ----------------
// Packed layout: tile (kt,nt) of 32x16 -> 256 u32; u32 index = tile*256 + lane*8 + j
// lane,j holds W[kt*32 + (lane>=16?16:0) + 2j + {lo,hi}][nt*16 + (lane&15)]
__global__ __launch_bounds__(256) void pack_w_bf16(const float* __restrict__ W,
                                                   unsigned int* __restrict__ Wp,
                                                   int K, int N) {
  size_t i = (size_t)blockIdx.x * 256 + threadIdx.x;
  size_t total = ((size_t)K * N) >> 1;
  if (i >= total) return;
  int Ntiles = N >> 4;
  size_t tile = i >> 8;
  int rem = (int)(i & 255);
  int lane = rem >> 3, j = rem & 7;
  int kt = (int)(tile / Ntiles), nt = (int)(tile % Ntiles);
  int n  = nt * 16 + (lane & 15);
  int k0 = kt * 32 + ((lane >> 4) << 4) + 2 * j;
  unsigned int lo = f2bf(W[(size_t)k0 * N + n]);
  unsigned int hi = f2bf(W[((size_t)k0 + 1) * N + n]);
  Wp[i] = lo | (hi << 16);
}

// ---------------- prep kernels (gathers -> bf16 activation rows) ----------------
__global__ __launch_bounds__(256) void prep_A0(const float* __restrict__ rel_t,
                                               const float* __restrict__ ent_t,
                                               const float* __restrict__ h0_prev,
                                               const int* __restrict__ prev_rel,
                                               const int* __restrict__ cur_ent,
                                               unsigned short* __restrict__ A0) {
  int idx = blockIdx.x * 256 + threadIdx.x;      // B*1024
  int b = idx >> 10, d = idx & 1023;
  float v;
  if (d < 256)       v = rel_t[(size_t)prev_rel[b] * 256 + d];
  else if (d < 512)  v = ent_t[(size_t)cur_ent[b] * 256 + (d - 256)];
  else               v = h0_prev[(size_t)b * 512 + (d - 512)];
  A0[(size_t)b * 1024 + d] = f2bf(v);
}

__global__ __launch_bounds__(256) void prep_hcols(const float* __restrict__ h,
                                                  unsigned short* __restrict__ A) {
  int idx = blockIdx.x * 256 + threadIdx.x;      // B*512
  int b = idx >> 9, d = idx & 511;
  A[(size_t)b * 1024 + 512 + d] = f2bf(h[(size_t)b * 512 + d]);
}

__global__ __launch_bounds__(256) void prep_sq_tail(const float* __restrict__ ent_t,
                                                    const int* __restrict__ cur_ent,
                                                    const float* __restrict__ query,
                                                    unsigned short* __restrict__ Asq) {
  int idx = blockIdx.x * 256 + threadIdx.x;      // B*512
  int b = idx >> 9, d = idx & 511;
  float v = (d < 256) ? ent_t[(size_t)cur_ent[b] * 256 + d]
                      : query[(size_t)b * 256 + (d - 256)];
  Asq[(size_t)b * 1024 + 512 + d] = f2bf(v);
}

// ---------------- WMMA bf16 GEMM with TDM-staged B slab in LDS ----------------
// C[M,N] = A[M,K] * W[K,N] + bias.  A: bf16 row-major (lda elems). Wp: packed fragments.
// Block = 8 waves = 128 M-rows x 64 N-cols. The block's B slab (Ktiles rows x 1024 u32)
// is DMA'd to LDS once by the Tensor Data Mover, then all waves read fragments via DS.
__global__ __launch_bounds__(256) void gemm_wmma_bf16(const unsigned short* __restrict__ A, int lda,
                                                      const unsigned int* __restrict__ Wp,
                                                      const float* __restrict__ bias,
                                                      float* __restrict__ C,
                                                      unsigned short* __restrict__ Cbf, int ldcbf,
                                                      int K, int N, int relu) {
  extern __shared__ __align__(16) unsigned int ldsB[];   // Ktiles * 1024 u32
  const int lane = threadIdx.x & 31;
  const int wave = threadIdx.x >> 5;
  const int m0 = blockIdx.y * 128 + wave * 16;
  const int n0 = blockIdx.x * 64;
  const int Ktiles = K >> 5;                             // always even (16 or 32)
  const int Nt16 = N >> 4;

  // ---- stage B slab: rows kt=0..Ktiles-1, each 4 tiles * 256 u32 contiguous ----
  const unsigned int* gsrc = Wp + (size_t)(n0 >> 4) * 256;   // tile (0, nt0)
#if __has_builtin(__builtin_amdgcn_tensor_load_to_lds)
  if (wave == 0) {
    unsigned long long ga = (unsigned long long)(uintptr_t)gsrc;
    unsigned int lds_addr = (unsigned int)(uintptr_t)&ldsB[0];
    v4u g0;
    g0.x = 1u;                                               // count=1, user mode
    g0.y = lds_addr;                                         // lds_addr (bytes)
    g0.z = (unsigned int)(ga & 0xFFFFFFFFu);                 // global_addr[31:0]
    g0.w = (unsigned int)((ga >> 32) & 0x01FFFFFFu) | (2u << 30);  // addr[56:32] | type=2
    v8i g1;
    g1[0] = (int)(2u << 16);                                 // data_size = 4B; no mask/pad/iter
    g1[1] = (int)0xFFFF0000u;                                // tensor_dim0 (large) bits 63:48
    g1[2] = (int)(0x0000FFFFu | ((unsigned)Ktiles << 16));   // dim0 hi | tensor_dim1 = Ktiles
    g1[3] = (int)(1024u << 16);                              // tile_dim0 = 1024 elements
    g1[4] = (int)(unsigned)Ktiles;                           // tile_dim1 = Ktiles, tile_dim2=0
    g1[5] = (int)(unsigned)(Nt16 * 256);                     // tensor_dim0_stride (elems)
    g1[6] = 0;
    g1[7] = 0;
    v4i gz = {0, 0, 0, 0};
#if __clang_major__ >= 23
    v8i gz8 = {0, 0, 0, 0, 0, 0, 0, 0};
    __builtin_amdgcn_tensor_load_to_lds(g0, g1, gz, gz, gz8, 0);
#else
    __builtin_amdgcn_tensor_load_to_lds(g0, g1, gz, gz, 0);
#endif
    __builtin_amdgcn_s_wait_tensorcnt(0);
  }
#else
  for (int i = threadIdx.x; i < Ktiles * 1024; i += 256)
    ldsB[i] = gsrc[(size_t)(i >> 10) * (Nt16 * 256) + (i & 1023)];
#endif
  __syncthreads();

  // ---- main loop, manually software-pipelined x2 with named A buffers (no spills) ----
  v8f acc[4] = {};
  const int mrow = m0 + (lane & 15);
  const int kh = (lane >> 4) << 4;                           // 0 or 16
  const unsigned short* arow = A + (size_t)mrow * lda + kh;

  AF a0, a1;
  a0.f[0] = *(const vf4*)(arow);
  a0.f[1] = *(const vf4*)(arow + 8);
  for (int kt = 0; kt < Ktiles; kt += 2) {
    // load A for kt+1 while kt's WMMAs run
    a1.f[0] = *(const vf4*)(arow + (kt + 1) * 32);
    a1.f[1] = *(const vf4*)(arow + (kt + 1) * 32 + 8);
    __builtin_prefetch(arow + (kt + 2) * 32, 0, 3);          // stream A
    wmma_row(ldsB + (size_t)kt * 1024 + lane * 8, a0.v, acc);
    // load A for kt+2 while kt+1's WMMAs run
    if (kt + 2 < Ktiles) {
      a0.f[0] = *(const vf4*)(arow + (kt + 2) * 32);
      a0.f[1] = *(const vf4*)(arow + (kt + 2) * 32 + 8);
    }
    wmma_row(ldsB + (size_t)(kt + 1) * 1024 + lane * 8, a1.v, acc);
  }

#pragma unroll
  for (int t = 0; t < 4; ++t) {
    int n = n0 + t * 16 + (lane & 15);
    float bv = bias ? bias[n] : 0.0f;
#pragma unroll
    for (int r = 0; r < 8; ++r) {
      int m = m0 + r + ((lane >> 4) << 3);
      float v = acc[t][r] + bv;
      if (relu) v = fmaxf(v, 0.0f);
      if (C)   C[(size_t)m * N + n] = v;
      if (Cbf) Cbf[(size_t)m * ldcbf + n] = f2bf(v);
    }
  }
}

// ---------------- peephole LSTM gates ----------------
__global__ __launch_bounds__(256) void lstm_gates(const float* __restrict__ z,
                                                  const float* __restrict__ c_prev,
                                                  const float* __restrict__ peep3,  // [3,512]
                                                  float* __restrict__ c_out,
                                                  float* __restrict__ h_out,
                                                  unsigned short* __restrict__ Anext, int acol) {
  int idx = blockIdx.x * 256 + threadIdx.x;     // B*512
  int b = idx >> 9, d = idx & 511;
  const float* zb = z + (size_t)b * 2048;
  float gi = zb[d], gj = zb[512 + d], gf = zb[1024 + d], go = zb[1536 + d];
  float c = c_prev[(size_t)b * 512 + d];
  float wi = peep3[d], wf = peep3[512 + d], wo = peep3[1024 + d];
  gi = sigm(gi + wi * c);
  gf = sigm(gf + 1.0f + wf * c);
  float cn = gf * c + gi * tanhf(gj);
  go = sigm(go + wo * cn);
  float hn = go * tanhf(cn);
  c_out[(size_t)b * 512 + d] = cn;
  h_out[(size_t)b * 512 + d] = hn;
  Anext[(size_t)b * 1024 + acol + d] = f2bf(hn);
}

// ---------------- candidate scores: prelim[b,a] = cand[b,a,:] . mlp[b,:] ----------------
__global__ __launch_bounds__(256) void scores_kernel(const float* __restrict__ rel_t,
                                                     const float* __restrict__ ent_t,
                                                     const int* __restrict__ next_rel,
                                                     const int* __restrict__ next_ent,
                                                     const float* __restrict__ mlp,
                                                     float* __restrict__ prelim,
                                                     float* __restrict__ masked) {
  __shared__ float m[512];
  int b = blockIdx.x;
  for (int d = threadIdx.x; d < 512; d += 256) m[d] = mlp[(size_t)b * 512 + d];
  __syncthreads();
  for (int a = threadIdx.x; a < AACT; a += 256) {
    int r = next_rel[(size_t)b * AACT + a];
    int e = next_ent[(size_t)b * AACT + a];
    const vf4* rp = (const vf4*)(rel_t + (size_t)r * 256);
    const vf4* ep = (const vf4*)(ent_t + (size_t)e * 256);
    float s = 0.0f;
#pragma unroll 4
    for (int q = 0; q < 64; ++q) {
      vf4 v = rp[q];
      s += v.x * m[4 * q] + v.y * m[4 * q + 1] + v.z * m[4 * q + 2] + v.w * m[4 * q + 3];
    }
#pragma unroll 4
    for (int q = 0; q < 64; ++q) {
      vf4 v = ep[q];
      s += v.x * m[256 + 4 * q] + v.y * m[256 + 4 * q + 1] + v.z * m[256 + 4 * q + 2] + v.w * m[256 + 4 * q + 3];
    }
    prelim[(size_t)b * AACT + a] = s;
    masked[(size_t)b * AACT + a] = (r == 0) ? NEGV : s;   // RPAD == 0
  }
}

// ---------------- softmax / Gumbel-max sample / loss (one wave32 per row) ----------------
__global__ __launch_bounds__(256) void sample_kernel(const float* __restrict__ masked,
                                                     const int* __restrict__ next_rel,
                                                     float* __restrict__ out) {
  int wave = threadIdx.x >> 5, lane = threadIdx.x & 31;
  int b = blockIdx.x * 8 + wave;
  if (b >= BSZ) return;
  const float* s = masked + (size_t)b * AACT;
  float sv[7];
  float smax = -3.4e38f;
#pragma unroll
  for (int i = 0; i < 7; ++i) {
    int a = lane + 32 * i;
    sv[i] = (a < AACT) ? s[a] : -3.4e38f;
    smax = fmaxf(smax, sv[i]);
  }
  for (int msk = 16; msk; msk >>= 1) smax = fmaxf(smax, __shfl_xor(smax, msk, 32));
  float sum = 0.0f;
#pragma unroll
  for (int i = 0; i < 7; ++i) {
    int a = lane + 32 * i;
    if (a < AACT) sum += __expf(sv[i] - smax);
  }
  for (int msk = 16; msk; msk >>= 1) sum += __shfl_xor(sum, msk, 32);
  float logZ = smax + __logf(sum);
#pragma unroll
  for (int i = 0; i < 7; ++i) {
    int a = lane + 32 * i;
    if (a < AACT) out[OUT_LP + (size_t)b * AACT + a] = sv[i] - logZ;
  }
  float bg = -3.4e38f; int ba = 0;
#pragma unroll
  for (int i = 0; i < 7; ++i) {
    int a = lane + 32 * i;
    if (a < AACT) {
      float u = rand01((unsigned int)(b * AACT + a) ^ 0x9E3779B9u);
      float g = sv[i] - __logf(-__logf(u));
      if (g > bg) { bg = g; ba = a; }
    }
  }
  for (int msk = 16; msk; msk >>= 1) {
    float og = __shfl_xor(bg, msk, 32);
    int   oa = __shfl_xor(ba, msk, 32);
    if (og > bg || (og == bg && oa < ba)) { bg = og; ba = oa; }
  }
  if (lane == 0) {
    out[OUT_LOSS + b] = logZ - s[ba];                               // -log_prob[idx]
    out[OUT_AIDX + b] = (float)ba;
    out[OUT_CREL + b] = (float)next_rel[(size_t)b * AACT + ba];
  }
}

// ---------------- host-side orchestration ----------------
extern "C" void kernel_launch(void* const* d_in, const int* in_sizes, int n_in,
                              void* d_out, int out_size, void* d_ws, size_t ws_size,
                              hipStream_t stream) {
  (void)in_sizes; (void)n_in; (void)out_size; (void)ws_size;
  const int*   next_rel   = (const int*)d_in[0];
  const int*   next_ent   = (const int*)d_in[1];
  const float* prev_state = (const float*)d_in[2];     // [2,2,B,512]
  const int*   prev_relid = (const int*)d_in[3];
  const float* query      = (const float*)d_in[4];
  const int*   cur_ent    = (const int*)d_in[5];
  const float* rel_t      = (const float*)d_in[7];
  const float* ent_t      = (const float*)d_in[8];
  const float* lstm_W     = (const float*)d_in[9];     // [2,1024,2048]
  const float* lstm_b     = (const float*)d_in[10];    // [2,2048]
  const float* peep       = (const float*)d_in[11];    // [2,3,512]
  const float* W1         = (const float*)d_in[12];    // [1024,512]
  const float* b1         = (const float*)d_in[13];
  const float* W2         = (const float*)d_in[14];    // [512,512]
  const float* b2         = (const float*)d_in[15];
  float* out = (float*)d_out;
  float* new_state = out + OUT_STATE;

  // workspace carve-up
  size_t off = 0;
  char* ws = (char*)d_ws;
  auto take = [&](size_t bytes) { void* p = ws + off; off += (bytes + 255) & ~(size_t)255; return p; };
  unsigned int*  wp_l0 = (unsigned int*) take((size_t)1024 * 2048 / 2 * 4);
  unsigned int*  wp_l1 = (unsigned int*) take((size_t)1024 * 2048 / 2 * 4);
  unsigned int*  wp_W1 = (unsigned int*) take((size_t)1024 * 512 / 2 * 4);
  unsigned int*  wp_W2 = (unsigned int*) take((size_t)512 * 512 / 2 * 4);
  unsigned short* A0   = (unsigned short*)take((size_t)BSZ * 1024 * 2);
  unsigned short* A1   = (unsigned short*)take((size_t)BSZ * 1024 * 2);
  unsigned short* Asq  = (unsigned short*)take((size_t)BSZ * 1024 * 2);
  unsigned short* hidb = (unsigned short*)take((size_t)BSZ * 512 * 2);
  float* z      = (float*)take((size_t)BSZ * 2048 * 4);
  float* mlp    = (float*)take((size_t)BSZ * 512 * 4);
  float* masked = (float*)take((size_t)BSZ * AACT * 4);

  // 1) pack weights -> bf16 fragment layout
  pack_w_bf16<<<(1024 * 2048 / 2 + 255) / 256, 256, 0, stream>>>(lstm_W, wp_l0, 1024, 2048);
  pack_w_bf16<<<(1024 * 2048 / 2 + 255) / 256, 256, 0, stream>>>(lstm_W + (size_t)1024 * 2048, wp_l1, 1024, 2048);
  pack_w_bf16<<<(1024 * 512 / 2 + 255) / 256, 256, 0, stream>>>(W1, wp_W1, 1024, 512);
  pack_w_bf16<<<(512 * 512 / 2 + 255) / 256, 256, 0, stream>>>(W2, wp_W2, 512, 512);

  // 2) gathers
  prep_A0<<<(BSZ * 1024) / 256, 256, 0, stream>>>(rel_t, ent_t, prev_state + (size_t)1 * BD,
                                                  prev_relid, cur_ent, A0);
  prep_hcols<<<(BSZ * 512) / 256, 256, 0, stream>>>(prev_state + (size_t)3 * BD, A1);
  prep_sq_tail<<<(BSZ * 512) / 256, 256, 0, stream>>>(ent_t, cur_ent, query, Asq);

  // 3) LSTM layer 0  (B slab: (1024/32)*4KB = 128KB LDS)
  dim3 gL(2048 / 64, BSZ / 128);
  gemm_wmma_bf16<<<gL, 256, (1024 / 32) * 4096, stream>>>(A0, 1024, wp_l0, lstm_b, z, nullptr, 0, 1024, 2048, 0);
  lstm_gates<<<(BSZ * 512) / 256, 256, 0, stream>>>(z, prev_state, peep,
                                                    new_state, new_state + BD, A1, 0);
  // 4) LSTM layer 1
  gemm_wmma_bf16<<<gL, 256, (1024 / 32) * 4096, stream>>>(A1, 1024, wp_l1, lstm_b + 2048, z, nullptr, 0, 1024, 2048, 0);
  lstm_gates<<<(BSZ * 512) / 256, 256, 0, stream>>>(z, prev_state + (size_t)2 * BD, peep + 3 * 512,
                                                    new_state + (size_t)2 * BD, new_state + (size_t)3 * BD,
                                                    Asq, 0);

  // 5) policy MLP
  dim3 gM(512 / 64, BSZ / 128);
  gemm_wmma_bf16<<<gM, 256, (1024 / 32) * 4096, stream>>>(Asq, 1024, wp_W1, b1, nullptr, hidb, 512, 1024, 512, 1);
  gemm_wmma_bf16<<<gM, 256, (512 / 32) * 4096, stream>>>(hidb, 512, wp_W2, b2, mlp, nullptr, 0, 512, 512, 1);

  // 6) candidate scores + mask
  scores_kernel<<<BSZ, 256, 0, stream>>>(rel_t, ent_t, next_rel, next_ent, mlp, out + OUT_PRELIM, masked);

  // 7) log_softmax, Gumbel-max sample, loss, chosen relation
  sample_kernel<<<BSZ / 8, 256, 0, stream>>>(masked, next_rel, out);
}
